// MOELayer_47579647705878
// MI455X (gfx1250) — matrix operand order repaired and checked by reading, not compiled, not run on hardware
//
#include <hip/hip_runtime.h>

#define S_TOK 8192
#define M_DIM 1024
#define E_NUM 8
#define H_DIM 4096
#define CAP   1024   // S_TOK / E_NUM, capacity_factor = 1.0

typedef __attribute__((ext_vector_type(8)))  __bf16 v8bf;
typedef __attribute__((ext_vector_type(16))) __bf16 v16bf;
typedef __attribute__((ext_vector_type(8)))  float  v8f;
typedef __attribute__((ext_vector_type(4)))  int    v4i;
typedef __attribute__((ext_vector_type(8)))  int    v8i;

__device__ __forceinline__ unsigned short bf16bits(float f) {
  return __builtin_bit_cast(unsigned short, (__bf16)f);
}
__device__ __forceinline__ unsigned pack_bf16(float a, float b) {
  return (unsigned)bf16bits(a) | ((unsigned)bf16bits(b) << 16);
}

// ---------------------------------------------------------------- TDM
// Tensor Data Mover load of a [rows x 32] bf16 tile (row stride = K elements
// in global memory) into LDS at lds_addr. 2D descriptor, 2 SGPR groups.
__device__ __forceinline__ void tdm_load_tile(unsigned lds_addr,
                                              const void* gaddr,
                                              int K, int rows) {
  unsigned long long ga = (unsigned long long)gaddr;
  v4i g0;
  g0[0] = 1;                                   // count=1, user descriptor
  g0[1] = (int)lds_addr;                       // LDS byte address
  g0[2] = (int)(ga & 0xFFFFFFFFu);             // global addr [31:0]
  g0[3] = (int)(((ga >> 32) & 0x01FFFFFFu)     // global addr [56:32]
                | 0x80000000u);                // type=2 ("image") in [127:126]
  v8i g1;
  g1[0] = 0x00010000;                          // wg_mask=0, data_size=1 (2B)
  g1[1] = (int)(((unsigned)K & 0xFFFFu) << 16);        // tensor_dim0 lo16
  g1[2] = (int)((((unsigned)K >> 16) & 0xFFFFu)        // tensor_dim0 hi16
                | ((unsigned)rows << 16));             // tensor_dim1 lo16
  g1[3] = (int)(32u << 16);                    // tensor_dim1 hi=0 | tile_dim0=32
  g1[4] = rows;                                // tile_dim1=rows, tile_dim2=0
  g1[5] = K;                                   // tensor_dim0_stride [31:0]
  g1[6] = 0;                                   // stride hi, dim1_stride lo
  g1[7] = 0;
  asm volatile("tensor_load_to_lds %0, %1" :: "s"(g0), "s"(g1) : "memory");
}

__device__ __forceinline__ unsigned lds_offset(const void* p) {
  return (unsigned)(size_t)p;   // low 32 bits of flat shared addr = LDS offset
}

// ---------------------------------------------------------------- init
__global__ void init_kernel(int* __restrict__ tfs, float* __restrict__ me_sum,
                            unsigned int* __restrict__ ce_cnt) {
  int i = blockIdx.x * blockDim.x + threadIdx.x;
  if (i < E_NUM * CAP) tfs[i] = -1;
  if (i < E_NUM) { me_sum[i] = 0.0f; ce_cnt[i] = 0u; }
}

// ---------------------------------------------------------------- wconv
// Per-expert transpose + f32->bf16: W[K,N] row-major -> WT[N,K] row-major.
// 64x64 tiles via padded LDS; packed 2xbf16 dword stores.
__launch_bounds__(256)
__global__ void wconv_kernel(const float* __restrict__ W,
                             unsigned short* __restrict__ WT,
                             int K, int N) {
  __shared__ float tile[64][65];
  const float*    We  = W  + (size_t)blockIdx.z * K * N;
  unsigned short* WTe = WT + (size_t)blockIdx.z * K * N;
  const int tk = blockIdx.y * 64;
  const int tn = blockIdx.x * 64;
  const int t  = threadIdx.x;
  {
    int r = t >> 2, c16 = (t & 3) * 16;
    const float4* g = (const float4*)(We + (size_t)(tk + r) * N + tn + c16);
#pragma unroll
    for (int j = 0; j < 4; ++j) {
      float4 v = g[j];
      tile[r][c16 + j * 4 + 0] = v.x;
      tile[r][c16 + j * 4 + 1] = v.y;
      tile[r][c16 + j * 4 + 2] = v.z;
      tile[r][c16 + j * 4 + 3] = v.w;
    }
  }
  __syncthreads();
  {
    int n = t >> 2, kc = (t & 3) * 16;
    unsigned o[8];
#pragma unroll
    for (int j = 0; j < 8; ++j)
      o[j] = pack_bf16(tile[kc + 2 * j][n], tile[kc + 2 * j + 1][n]);
    uint4* dst = (uint4*)(WTe + (size_t)(tn + n) * K + tk + kc);
    uint4 d0; d0.x = o[0]; d0.y = o[1]; d0.z = o[2]; d0.w = o[3];
    uint4 d1; d1.x = o[4]; d1.y = o[5]; d1.z = o[6]; d1.w = o[7];
    dst[0] = d0;
    dst[1] = d1;
  }
}

// ---------------------------------------------------------------- gate
__launch_bounds__(256)
__global__ void gate_kernel(const float* __restrict__ x,
                            const float* __restrict__ wg,
                            int* __restrict__ eidx,
                            float* __restrict__ gatev,
                            float* __restrict__ me_sum,
                            unsigned int* __restrict__ ce_cnt) {
  const int lane = threadIdx.x & 31;
  const int wv   = threadIdx.x >> 5;
  const int s    = blockIdx.x * 8 + wv;

  float acc[E_NUM];
#pragma unroll
  for (int e = 0; e < E_NUM; ++e) acc[e] = 0.0f;

  const float* xr = x + (size_t)s * M_DIM;
  for (int i = lane; i < M_DIM; i += 32) {
    float xv = xr[i];
    const float* wr = wg + i * E_NUM;
#pragma unroll
    for (int e = 0; e < E_NUM; ++e) acc[e] += xv * wr[e];
  }
#pragma unroll
  for (int off = 16; off > 0; off >>= 1) {
#pragma unroll
    for (int e = 0; e < E_NUM; ++e) acc[e] += __shfl_xor(acc[e], off, 32);
  }
  if (lane == 0) {
    float mx = acc[0];
#pragma unroll
    for (int e = 1; e < E_NUM; ++e) mx = fmaxf(mx, acc[e]);
    float p[E_NUM];
    float sum = 0.0f;
#pragma unroll
    for (int e = 0; e < E_NUM; ++e) { p[e] = __expf(acc[e] - mx); sum += p[e]; }
    float inv = 1.0f / sum;
    int arg = 0; float best = p[0];
#pragma unroll
    for (int e = 1; e < E_NUM; ++e) if (p[e] > best) { best = p[e]; arg = e; }
    eidx[s]  = arg;
    gatev[s] = best * inv;
#pragma unroll
    for (int e = 0; e < E_NUM; ++e) atomicAdd(&me_sum[e], p[e] * inv);
    atomicAdd(&ce_cnt[arg], 1u);
  }
}

// ---------------------------------------------------------------- assign
__global__ void assign_kernel(const int* __restrict__ eidx,
                              int* __restrict__ slot,
                              int* __restrict__ tfs,
                              const float* __restrict__ me_sum,
                              const unsigned int* __restrict__ ce_cnt,
                              float* __restrict__ laux) {
  const int lane = threadIdx.x;
  int count[E_NUM];
#pragma unroll
  for (int e = 0; e < E_NUM; ++e) count[e] = 0;
  const unsigned lower = (1u << lane) - 1u;

  for (int s0 = 0; s0 < S_TOK; s0 += 32) {
    int e = eidx[s0 + lane];
    int myslot = -1;
#pragma unroll
    for (int ee = 0; ee < E_NUM; ++ee) {
      unsigned m = (unsigned)__ballot(e == ee);
      if (e == ee) {
        int pos = count[ee] + __popc(m & lower);
        if (pos < CAP) {
          myslot = pos;
          tfs[ee * CAP + pos] = s0 + lane;
        }
      }
      count[ee] += __popc(m);
    }
    slot[s0 + lane] = myslot;
  }
  if (lane == 0) {
    float l = 0.0f;
#pragma unroll
    for (int e = 0; e < E_NUM; ++e)
      l += (me_sum[e] / (float)S_TOK) * ((float)ce_cnt[e] / (float)S_TOK);
    laux[0] = l * (float)E_NUM;
  }
}

// ---------------------------------------------------------------- gather
__global__ void gather_kernel(const float* __restrict__ x,
                              const int* __restrict__ tfs,
                              unsigned short* __restrict__ disp) {
  int slotid = blockIdx.x;
  int s = tfs[slotid];
  int m = threadIdx.x * 4;
  unsigned int lo = 0u, hi = 0u;
  if (s >= 0) {
    float4 v = *(const float4*)(x + (size_t)s * M_DIM + m);
    lo = pack_bf16(v.x, v.y);
    hi = pack_bf16(v.z, v.w);
  }
  uint2 w; w.x = lo; w.y = hi;
  *(uint2*)(disp + (size_t)slotid * M_DIM + m) = w;
}

// ---------------------------------------------------------------- GEMM
// C[Mo,N] = A[Mo,K](bf16) * BT[N,K](bf16, pre-transposed) per expert.
// 8 waves, 128x256 block tile, 64x64 wave tile (4x4 WMMA frags), K-step 32.
// BOTH tiles staged by the Tensor Data Mover (A by wave0, B by wave1, on
// independent TENSORcnt streams). Waves only do ds_load_b128 + WMMA.
template <bool RELU_BF16_OUT>
__launch_bounds__(256)
__global__ void gemm_kernel(const unsigned short* __restrict__ Aall,
                            const unsigned short* __restrict__ BTall,
                            unsigned short* __restrict__ OutBf,
                            float* __restrict__ OutF,
                            int Mo, int N, int K) {
  __shared__ __bf16 At[2][128][32];   // [row][k]          16 KB
  __shared__ __bf16 Bt[2][256][32];   // [n][k]            32 KB

  const unsigned short* A_e = Aall  + (size_t)blockIdx.z * Mo * K;
  const unsigned short* B_e = BTall + (size_t)blockIdx.z * N * K;
  const int mrow0 = blockIdx.y * 128;
  const int ncol0 = blockIdx.x * 256;

  const int t    = threadIdx.x;
  const int lane = t & 31;
  const int wv   = t >> 5;
  const int wr0  = (wv >> 2) * 64;   // wave row origin
  const int wc0  = (wv & 3) * 64;    // wave col origin
  const int l16  = lane & 15;
  const int lh   = lane >> 4;

  v8f acc[4][4];
#pragma unroll
  for (int r = 0; r < 4; ++r)
#pragma unroll
    for (int c = 0; c < 4; ++c) acc[r][c] = 0.0f;

  const unsigned short* Atile0 = A_e + (size_t)mrow0 * K;   // [128 x K]
  const unsigned short* Btile0 = B_e + (size_t)ncol0 * K;   // [256 x K]

  auto issue = [&](int buf, int k0) {
    if (t == 0) {
      tdm_load_tile(lds_offset(&At[buf][0][0]), Atile0 + k0, K, 128);
      if (k0 + 64 < K) __builtin_prefetch(Atile0 + k0 + 64, 0, 1);
    } else if (t == 32) {
      tdm_load_tile(lds_offset(&Bt[buf][0][0]), Btile0 + k0, K, 256);
      if (k0 + 64 < K) __builtin_prefetch(Btile0 + k0 + 64, 0, 1);
    }
  };

  // Prologue
  issue(0, 0);
  if (t == 0 || t == 32) __builtin_amdgcn_s_wait_tensorcnt(0);
  __syncthreads();

  const int ksteps = K >> 5;
  for (int kt = 0; kt < ksteps; ++kt) {
    const int cur = kt & 1;
    if (kt + 1 < ksteps) issue(cur ^ 1, (kt + 1) << 5);

    // A fragment: lane (l16,lh) -> row = l16, K = {lh*8+0..7, lh*8+16..23}
    v16bf afrag[4];
#pragma unroll
    for (int r = 0; r < 4; ++r) {
      const v8bf* p0 = (const v8bf*)&At[cur][wr0 + r * 16 + l16][lh * 8];
      const v8bf* p1 = (const v8bf*)&At[cur][wr0 + r * 16 + l16][lh * 8 + 16];
      v8bf lo = *p0, hi = *p1;
#pragma unroll
      for (int i = 0; i < 8; ++i) { afrag[r][i] = lo[i]; afrag[r][i + 8] = hi[i]; }
    }
    // B fragment: lane -> column n = l16, K = {lh*8+0..7, lh*8+16..23}
    v16bf bfrag[4];
#pragma unroll
    for (int c = 0; c < 4; ++c) {
      v8bf blo = *(const v8bf*)&Bt[cur][wc0 + c * 16 + l16][lh * 8];
      v8bf bhi = *(const v8bf*)&Bt[cur][wc0 + c * 16 + l16][lh * 8 + 16];
#pragma unroll
      for (int i = 0; i < 8; ++i) { bfrag[c][i] = blo[i]; bfrag[c][i + 8] = bhi[i]; }
    }

#pragma unroll
    for (int r = 0; r < 4; ++r)
#pragma unroll
      for (int c = 0; c < 4; ++c)
        acc[r][c] = __builtin_amdgcn_wmma_f32_16x16x32_bf16(
            false, afrag[r], false, bfrag[c], (short)0, acc[r][c], false, false);

    if (t == 0 || t == 32) __builtin_amdgcn_s_wait_tensorcnt(0);
    __syncthreads();
  }

  // Epilogue. C/D layout: VGPR g -> row = g + 8*lh, col = l16.
  const size_t obase = (size_t)blockIdx.z * Mo * N;
#pragma unroll
  for (int r = 0; r < 4; ++r) {
#pragma unroll
    for (int c = 0; c < 4; ++c) {
#pragma unroll
      for (int g = 0; g < 8; ++g) {
        int row = mrow0 + wr0 + r * 16 + lh * 8 + g;
        int col = ncol0 + wc0 + c * 16 + l16;
        float v = acc[r][c][g];
        if (RELU_BF16_OUT) {
          OutBf[obase + (size_t)row * N + col] = bf16bits(v > 0.0f ? v : 0.0f);
        } else {
          OutF[obase + (size_t)row * N + col] = v;
        }
      }
    }
  }
}

// ---------------------------------------------------------------- combine
__global__ void combine_kernel(const float* __restrict__ eo,
                               const int* __restrict__ slot,
                               const int* __restrict__ eidx,
                               const float* __restrict__ gatev,
                               float* __restrict__ out) {
  int s = blockIdx.x;
  int c = slot[s];
  int m = threadIdx.x * 4;
  float4 r; r.x = r.y = r.z = r.w = 0.0f;
  if (c >= 0) {
    int e   = eidx[s];
    float g = gatev[s];
    float4 v = *(const float4*)(eo + (size_t)(e * CAP + c) * M_DIM + m);
    r.x = g * v.x; r.y = g * v.y; r.z = g * v.z; r.w = g * v.w;
  }
  *(float4*)(out + (size_t)s * M_DIM + m) = r;
}

// ---------------------------------------------------------------- launch
extern "C" void kernel_launch(void* const* d_in, const int* in_sizes, int n_in,
                              void* d_out, int out_size, void* d_ws, size_t ws_size,
                              hipStream_t stream) {
  (void)in_sizes; (void)n_in; (void)out_size; (void)ws_size;
  const float* x  = (const float*)d_in[0];   // [8192,1024]
  const float* wg = (const float*)d_in[1];   // [1024,8]
  const float* w1 = (const float*)d_in[2];   // [8,1024,4096]
  const float* w2 = (const float*)d_in[3];   // [8,4096,1024]
  float* out = (float*)d_out;                // [8192*1024] out + [1] l_aux

  char* p = (char*)d_ws;
  size_t off = 0;
  auto take = [&](size_t bytes) {
    char* r = p + off;
    off = (off + bytes + 255) & ~(size_t)255;
    return r;
  };
  int*            eidx   = (int*)take((size_t)S_TOK * 4);
  float*          gatev  = (float*)take((size_t)S_TOK * 4);
  int*            slotv  = (int*)take((size_t)S_TOK * 4);
  int*            tfs    = (int*)take((size_t)E_NUM * CAP * 4);
  float*          me_sum = (float*)take((size_t)E_NUM * 4);
  unsigned int*   ce_cnt = (unsigned int*)take((size_t)E_NUM * 4);
  unsigned short* disp   = (unsigned short*)take((size_t)E_NUM * CAP * M_DIM * 2);    // 16 MB
  unsigned short* hbuf   = (unsigned short*)take((size_t)E_NUM * CAP * H_DIM * 2);    // 64 MB
  float*          eo     = (float*)take((size_t)E_NUM * CAP * M_DIM * 4);             // 32 MB
  unsigned short* w1T    = (unsigned short*)take((size_t)E_NUM * M_DIM * H_DIM * 2);  // 64 MB
  unsigned short* w2T    = (unsigned short*)take((size_t)E_NUM * M_DIM * H_DIM * 2);  // 64 MB

  init_kernel<<<(E_NUM * CAP + 255) / 256, 256, 0, stream>>>(tfs, me_sum, ce_cnt);

  // Pre-transpose + convert weights: w1[K=M,N=H] -> w1T[H,M]; w2[K=H,N=M] -> w2T[M,H]
  wconv_kernel<<<dim3(H_DIM / 64, M_DIM / 64, E_NUM), 256, 0, stream>>>(
      w1, w1T, M_DIM, H_DIM);
  wconv_kernel<<<dim3(M_DIM / 64, H_DIM / 64, E_NUM), 256, 0, stream>>>(
      w2, w2T, H_DIM, M_DIM);

  gate_kernel<<<S_TOK / 8, 256, 0, stream>>>(x, wg, eidx, gatev, me_sum, ce_cnt);
  assign_kernel<<<1, 32, 0, stream>>>(eidx, slotv, tfs, me_sum, ce_cnt,
                                      out + (size_t)S_TOK * M_DIM);
  gather_kernel<<<E_NUM * CAP, 256, 0, stream>>>(x, tfs, disp);

  // GEMM1: h = relu(disp @ w1), per expert 1024x4096, K=1024
  gemm_kernel<true><<<dim3(H_DIM / 256, CAP / 128, E_NUM), 256, 0, stream>>>(
      disp, w1T, hbuf, nullptr, CAP, H_DIM, M_DIM);
  // GEMM2: eo = h @ w2, per expert 1024x1024, K=4096
  gemm_kernel<false><<<dim3(M_DIM / 256, CAP / 128, E_NUM), 256, 0, stream>>>(
      hbuf, w2T, nullptr, eo, CAP, M_DIM, H_DIM);

  combine_kernel<<<S_TOK, 256, 0, stream>>>(eo, slotv, eidx, gatev, out);
}
